// Qwen3Attention_69758858822255
// MI455X (gfx1250) — matrix operand order repaired and checked by reading, not compiled, not run on hardware
//
#include <hip/hip_runtime.h>

// ---------------------------------------------------------------------------
// Qwen3 attention for MI455X (gfx1250): bf16 WMMA everywhere, flash-attention
// so S x S scores stay on-chip, double-buffered async global->LDS staging for
// GEMM B tiles (copy for K-step k+1 overlaps WMMAs of step k).
// ---------------------------------------------------------------------------

static constexpr int B_  = 2;
static constexpr int S_  = 2048;
static constexpr int D_  = 2048;
static constexpr int H_  = 16;
static constexpr int KV_ = 8;
static constexpr int HD_ = 128;
static constexpr int M_  = B_ * S_;          // 4096 (b,s) rows
static constexpr float EPS_   = 1e-6f;
static constexpr float SCALE_ = 0.08838834764831845f; // 128^-0.5

typedef __attribute__((ext_vector_type(16))) __bf16 v16bf;
typedef __attribute__((ext_vector_type(8)))  float  v8f;

__device__ __forceinline__ unsigned short f2bf(float f) {
  unsigned u = __float_as_uint(f);
  u += 0x7fffu + ((u >> 16) & 1u);
  return (unsigned short)(u >> 16);
}

__device__ __forceinline__ v8f wmma_bf16(v16bf a, v16bf b, v8f c) {
  // D = A(16x32) * B(32x16) + C, f32 accumulate
  return __builtin_amdgcn_wmma_f32_16x16x32_bf16(false, a, false, b, (short)0, c,
                                                 false, false);
}

__device__ __forceinline__ v16bf ld_frag(const unsigned short* p) {
  return *reinterpret_cast<const v16bf*>(p);
}

// ---------------------------------------------------------------------------
// Pack fp32 row-major (Mdim x Kdim) into bf16 A-operand fragments.
// Frag = 16 rows x 32 cols, 512 bf16, laid out in this file's contiguous-k A order:
//   lane<16 : row=lane,   k = 0..15;  lane>=16 : row=lane-16, k = 16..31
// ---------------------------------------------------------------------------
__global__ __launch_bounds__(256) void pack_a_kernel(const float* __restrict__ src,
                                                     unsigned short* __restrict__ dst,
                                                     int Mdim, int Kdim) {
  const int t = blockIdx.x * 256 + threadIdx.x;
  const int frag = t >> 5, lane = t & 31;
  const int kchunks = Kdim >> 5;
  const int mtile = frag / kchunks, kc = frag % kchunks;
  const int r = lane & 15, half = lane >> 4;
  const float* s = src + (size_t)(mtile * 16 + r) * Kdim + kc * 32;
  unsigned short tmp[16];
#pragma unroll
  for (int i = 0; i < 16; ++i) {
    const int dd = i + half * 16;
    tmp[i] = f2bf(s[dd]);
  }
  uint4* d = reinterpret_cast<uint4*>(dst + (size_t)frag * 512 + lane * 16);
  d[0] = *reinterpret_cast<uint4*>(&tmp[0]);
  d[1] = *reinterpret_cast<uint4*>(&tmp[8]);
}

// ---------------------------------------------------------------------------
// Pack fp32 row-major (Kdim x Ndim) weights into bf16 B-operand fragments.
// Frag = 32 rows(K) x 16 cols(N):
//   lane<16 : col=lane,   k=0..15 ; lane>=16 : col=lane-16, k=16..31
// Frag index order: [kchunk][ntile]  (8 consecutive ntiles -> contiguous 8KB)
// ---------------------------------------------------------------------------
__global__ __launch_bounds__(256) void pack_b_kernel(const float* __restrict__ src,
                                                     unsigned short* __restrict__ dst,
                                                     int Kdim, int Ndim) {
  const int t = blockIdx.x * 256 + threadIdx.x;
  const int frag = t >> 5, lane = t & 31;
  const int ntiles = Ndim >> 4;
  const int kc = frag / ntiles, nt = frag % ntiles;
  const int col = lane & 15, half = lane >> 4;
  const float* s = src + (size_t)(kc * 32 + half * 16) * Ndim + nt * 16 + col;
  unsigned short tmp[16];
#pragma unroll
  for (int i = 0; i < 16; ++i) tmp[i] = f2bf(s[(size_t)i * Ndim]);
  uint4* d = reinterpret_cast<uint4*>(dst + (size_t)frag * 512 + lane * 16);
  d[0] = *reinterpret_cast<uint4*>(&tmp[0]);
  d[1] = *reinterpret_cast<uint4*>(&tmp[8]);
}

// ---------------------------------------------------------------------------
// GEMM: C(Mdim x Ndim, f32) = A(packed bf16) * B(packed bf16)
// Block = 8 waves as 4(M) x 2(N). Block tile = 128(M) x 128(N).
// Wave tile = 32(M) x 64(N): 2 A-frags x 4 B-frags -> 8 WMMA per K-step.
// B fragments (8KB per K-step) double-buffered in LDS: wave 0 issues async
// global->LDS copies for step k+1 while all waves compute step k.
// ---------------------------------------------------------------------------
__global__ __launch_bounds__(256) void gemm_kernel(const unsigned short* __restrict__ Ap,
                                                   const unsigned short* __restrict__ Bp,
                                                   float* __restrict__ C,
                                                   int Mdim, int Ndim, int Kdim) {
  __shared__ __align__(16) unsigned short bsh[2][8 * 512]; // 2 x 8KB
  const int tid = threadIdx.x;
  const int wave = tid >> 5, lane = tid & 31;
  const int mquad = wave & 3;        // 0..3 -> 32-row strip
  const int nhalf = wave >> 2;       // 0..1 -> 64-col strip
  const int mtile0 = blockIdx.x * 8 + mquad * 2;
  const int nt0 = blockIdx.y * 8;    // first of 8 B tiles staged per block
  const int kchunks = Kdim >> 5, ntiles = Ndim >> 4;
  const unsigned short* arow0 = Ap + (size_t)mtile0 * kchunks * 512 + lane * 16;
  const unsigned short* arow1 = arow0 + (size_t)kchunks * 512;

  v8f acc0[4], acc1[4];
#pragma unroll
  for (int t = 0; t < 4; ++t) { acc0[t] = v8f{}; acc1[t] = v8f{}; }

  // prologue: stage K-step 0 into buffer 0
  if (tid < 32) {
    const unsigned short* gsrc = Bp + (size_t)nt0 * 512;
#pragma unroll
    for (int c = 0; c < 16; ++c) {
      const void* g = (const void*)(gsrc + c * 256 + lane * 8); // 16B per lane
      unsigned loff = (unsigned)(unsigned long long)(&bsh[0][c * 256 + lane * 8]);
      asm volatile("global_load_async_to_lds_b128 %0, %1, off"
                   :: "v"(loff), "v"(g) : "memory");
    }
  }

  for (int kc = 0; kc < kchunks; ++kc) {
    const int cur = kc & 1;
    if (tid < 32) asm volatile("s_wait_asynccnt 0" ::: "memory");
    __syncthreads(); // buffer `cur` filled; prior reads of `1-cur` long done
    if (kc + 1 < kchunks && tid < 32) {
      const unsigned short* gsrc = Bp + ((size_t)(kc + 1) * ntiles + nt0) * 512;
#pragma unroll
      for (int c = 0; c < 16; ++c) {
        const void* g = (const void*)(gsrc + c * 256 + lane * 8);
        unsigned loff =
            (unsigned)(unsigned long long)(&bsh[1 - cur][c * 256 + lane * 8]);
        asm volatile("global_load_async_to_lds_b128 %0, %1, off"
                     :: "v"(loff), "v"(g) : "memory");
      }
    }
    const v16bf a0 = ld_frag(arow0 + (size_t)kc * 512);
    const v16bf a1 = ld_frag(arow1 + (size_t)kc * 512);
#pragma unroll
    for (int t = 0; t < 4; ++t) {
      const v16bf b = *reinterpret_cast<const v16bf*>(
          &bsh[cur][(nhalf * 4 + t) * 512 + lane * 16]);
      acc0[t] = wmma_bf16(a0, b, acc0[t]);
      acc1[t] = wmma_bf16(a1, b, acc1[t]);
    }
  }

  // epilogue: C layout -> row-major f32
  const int colb = lane & 15;
  const int rhalf = (lane < 16) ? 0 : 8;
#pragma unroll
  for (int t = 0; t < 4; ++t) {
    const int col = (nt0 + nhalf * 4 + t) * 16 + colb;
#pragma unroll
    for (int g = 0; g < 8; ++g) {
      C[(size_t)(mtile0 * 16 + rhalf + g) * Ndim + col] = acc0[t][g];
      C[(size_t)((mtile0 + 1) * 16 + rhalf + g) * Ndim + col] = acc1[t][g];
    }
  }
}

// ---------------------------------------------------------------------------
// RMSNorm + RoPE for Q, writing bf16 directly in A-operand fragment layout.
// One wave per (b,s,h) row of 128 dims; 4 dims per lane.
// ---------------------------------------------------------------------------
__global__ __launch_bounds__(256) void qnorm_rope_pack(const float* __restrict__ qsrc,
                                                       const float* __restrict__ wn,
                                                       const float* __restrict__ cosT,
                                                       const float* __restrict__ sinT,
                                                       unsigned short* __restrict__ Qp) {
  const int lane = threadIdx.x & 31;
  const int row = blockIdx.x * 8 + (threadIdx.x >> 5); // (b*S+s)*H + h
  const int h = row & 15;
  const int s = (row >> 4) & (S_ - 1);
  const int b = row >> 15;
  const int d0 = lane * 4;
  float x[4];
  { float4 t = *reinterpret_cast<const float4*>(qsrc + (size_t)row * HD_ + d0);
    x[0] = t.x; x[1] = t.y; x[2] = t.z; x[3] = t.w; }
  float ss = x[0]*x[0] + x[1]*x[1] + x[2]*x[2] + x[3]*x[3];
#pragma unroll
  for (int off = 1; off < 32; off <<= 1) ss += __shfl_xor(ss, off, 32);
  const float r = rsqrtf(ss * (1.f / HD_) + EPS_);
  float w[4];
  { float4 t = *reinterpret_cast<const float4*>(wn + d0);
    w[0] = t.x; w[1] = t.y; w[2] = t.z; w[3] = t.w; }
  float a[4], pa[4];
#pragma unroll
  for (int j = 0; j < 4; ++j) a[j] = x[j] * r * w[j];
#pragma unroll
  for (int j = 0; j < 4; ++j) pa[j] = __shfl_xor(a[j], 16, 32); // d +/- 64 partner
  float cs[4], sn[4];
  { float4 t = *reinterpret_cast<const float4*>(cosT + (size_t)s * HD_ + d0);
    cs[0] = t.x; cs[1] = t.y; cs[2] = t.z; cs[3] = t.w; }
  { float4 t = *reinterpret_cast<const float4*>(sinT + (size_t)s * HD_ + d0);
    sn[0] = t.x; sn[1] = t.y; sn[2] = t.z; sn[3] = t.w; }
  union { unsigned short s4[4]; uint2 u; } t;
#pragma unroll
  for (int j = 0; j < 4; ++j) {
    const float o = (lane < 16) ? (a[j] * cs[j] - pa[j] * sn[j])
                                : (a[j] * cs[j] + pa[j] * sn[j]);
    t.s4[j] = f2bf(o);
  }
  const int stile = s >> 4, rpos = s & 15;
  const int cchunk = d0 >> 5, dd0 = d0 & 31;
  const int hf = (dd0 & 16) ? 1 : 0;
  const int lanep = rpos + 16 * hf;
  const int idx0 = dd0 & 15;
  unsigned short* dst = Qp + ((((size_t)(b * H_ + h) * (S_ / 16) + stile) * 4 + cchunk) * 512)
                        + lanep * 16 + idx0;
  *reinterpret_cast<uint2*>(dst) = t.u;
}

// ---------------------------------------------------------------------------
// RMSNorm + RoPE for K, writing bf16 in B-operand (K^T) fragment layout:
// frag = (hd-chunk 32 rows) x (16 keys). lane' = key%16 + 16*(hd%32>=16).
// ---------------------------------------------------------------------------
__global__ __launch_bounds__(256) void knorm_rope_pack(const float* __restrict__ ksrc,
                                                       const float* __restrict__ wn,
                                                       const float* __restrict__ cosT,
                                                       const float* __restrict__ sinT,
                                                       unsigned short* __restrict__ Kp) {
  const int lane = threadIdx.x & 31;
  const int row = blockIdx.x * 8 + (threadIdx.x >> 5); // (b*S+s)*KV + kvh
  const int kvh = row & 7;
  const int s = (row >> 3) & (S_ - 1);
  const int b = row >> 14;
  const int d0 = lane * 4;
  float x[4];
  { float4 t = *reinterpret_cast<const float4*>(ksrc + (size_t)row * HD_ + d0);
    x[0] = t.x; x[1] = t.y; x[2] = t.z; x[3] = t.w; }
  float ss = x[0]*x[0] + x[1]*x[1] + x[2]*x[2] + x[3]*x[3];
#pragma unroll
  for (int off = 1; off < 32; off <<= 1) ss += __shfl_xor(ss, off, 32);
  const float r = rsqrtf(ss * (1.f / HD_) + EPS_);
  float w[4];
  { float4 t = *reinterpret_cast<const float4*>(wn + d0);
    w[0] = t.x; w[1] = t.y; w[2] = t.z; w[3] = t.w; }
  float a[4], pa[4];
#pragma unroll
  for (int j = 0; j < 4; ++j) a[j] = x[j] * r * w[j];
#pragma unroll
  for (int j = 0; j < 4; ++j) pa[j] = __shfl_xor(a[j], 16, 32);
  float cs[4], sn[4];
  { float4 t = *reinterpret_cast<const float4*>(cosT + (size_t)s * HD_ + d0);
    cs[0] = t.x; cs[1] = t.y; cs[2] = t.z; cs[3] = t.w; }
  { float4 t = *reinterpret_cast<const float4*>(sinT + (size_t)s * HD_ + d0);
    sn[0] = t.x; sn[1] = t.y; sn[2] = t.z; sn[3] = t.w; }
  union { unsigned short s4[4]; uint2 u; } t;
#pragma unroll
  for (int j = 0; j < 4; ++j) {
    const float o = (lane < 16) ? (a[j] * cs[j] - pa[j] * sn[j])
                                : (a[j] * cs[j] + pa[j] * sn[j]);
    t.s4[j] = f2bf(o);
  }
  const int stile = s >> 4, rpos = s & 15;
  const int cchunk = d0 >> 5, dd0 = d0 & 31;
  const int lanep = rpos + 16 * ((dd0 & 16) ? 1 : 0);
  const int idx0 = dd0 & 15;
  unsigned short* dst = Kp + ((((size_t)(b * KV_ + kvh) * (S_ / 16) + stile) * 4 + cchunk) * 512)
                        + lanep * 16 + idx0;
  *reinterpret_cast<uint2*>(dst) = t.u;
}

// ---------------------------------------------------------------------------
// V -> bf16 B-operand fragments for P*V: frag = (32 keys) x (16 hd cols).
// ---------------------------------------------------------------------------
__global__ __launch_bounds__(256) void vpack_kernel(const float* __restrict__ vsrc,
                                                    unsigned short* __restrict__ Vp) {
  const int lane = threadIdx.x & 31;
  const int row = blockIdx.x * 8 + (threadIdx.x >> 5); // (b*S+s)*KV + kvh
  const int kvh = row & 7;
  const int s = (row >> 3) & (S_ - 1);
  const int b = row >> 14;
  const int d0 = lane * 4;
  float4 t = *reinterpret_cast<const float4*>(vsrc + (size_t)row * HD_ + d0);
  float x[4] = {t.x, t.y, t.z, t.w};
  const int sc = s >> 5, krow = s & 31;
  const int hf = krow >> 4, idx = krow & 15;
  unsigned short* dbase =
      Vp + ((((size_t)(b * KV_ + kvh) * (S_ / 32) + sc) * 8 + (d0 >> 4)) * 512);
#pragma unroll
  for (int j = 0; j < 4; ++j) {
    const int col = (d0 + j) & 15;
    dbase[(col + 16 * hf) * 16 + idx] = f2bf(x[j]);
  }
}

// ---------------------------------------------------------------------------
// Flash attention: one wave per (b,h,16-row q tile). Per 32 keys:
//   8 WMMA (Q*K^T) -> online softmax (shfl butterflies) ->
//   P C-layout -> A-layout transpose through LDS -> 8 WMMA (P*V).
// Output written f32 row-major (b*S+s, h*128+d) for the Wo GEMM pack.
// ---------------------------------------------------------------------------
__global__ __launch_bounds__(32) void flash_kernel(const unsigned short* __restrict__ Qp,
                                                   const unsigned short* __restrict__ Kp,
                                                   const unsigned short* __restrict__ Vp,
                                                   float* __restrict__ attnO) {
  __shared__ __align__(16) unsigned short plds[16 * 32];
  const int lane = threadIdx.x;
  const int bh = blockIdx.x;
  const int b = bh >> 4, h = bh & 15;
  const int qi = blockIdx.y;
  const int kvh = h >> 1; // GQA: H/KV = 2
  const int coln = lane & 15;
  const int half = lane >> 4;
  const int rr = half * 8;

  v16bf qf[4];
  {
    const unsigned short* qb =
        Qp + ((((size_t)(b * H_ + h) * (S_ / 16) + qi) * 4) * 512) + lane * 16;
#pragma unroll
    for (int c = 0; c < 4; ++c) qf[c] = ld_frag(qb + c * 512);
  }

  v8f o[8];
#pragma unroll
  for (int n = 0; n < 8; ++n) o[n] = v8f{};
  float mrow[8], lrow[8];
#pragma unroll
  for (int g = 0; g < 8; ++g) { mrow[g] = -1e30f; lrow[g] = 0.f; }

  const int njj = (qi * 16 + 47) >> 5; // ceil((qi*16+16)/32)
  for (int jj = 0; jj < njj; ++jj) {
    const int base = jj * 32;
    v8f s0 = v8f{}, s1 = v8f{};
    {
      const unsigned short* kb =
          Kp + ((((size_t)(b * KV_ + kvh) * (S_ / 16) + 2 * jj) * 4) * 512) + lane * 16;
#pragma unroll
      for (int c = 0; c < 4; ++c) s0 = wmma_bf16(qf[c], ld_frag(kb + c * 512), s0);
      kb += 4 * 512;
#pragma unroll
      for (int c = 0; c < 4; ++c) s1 = wmma_bf16(qf[c], ld_frag(kb + c * 512), s1);
    }
    float p0[8], p1[8], alpha[8];
#pragma unroll
    for (int g = 0; g < 8; ++g) {
      const int qrow = qi * 16 + rr + g;
      float a = s0[g] * SCALE_;  if (base + coln > qrow)      a = -1e30f;
      float c2 = s1[g] * SCALE_; if (base + 16 + coln > qrow) c2 = -1e30f;
      float mn = fmaxf(a, c2);
#pragma unroll
      for (int off = 1; off < 16; off <<= 1) mn = fmaxf(mn, __shfl_xor(mn, off, 32));
      const float mnew = fmaxf(mrow[g], mn);
      alpha[g] = __expf(mrow[g] - mnew);
      mrow[g] = mnew;
      const float e0 = __expf(a - mnew), e1 = __expf(c2 - mnew);
      p0[g] = e0; p1[g] = e1;
      float ls = e0 + e1;
#pragma unroll
      for (int off = 1; off < 16; off <<= 1) ls += __shfl_xor(ls, off, 32);
      lrow[g] = lrow[g] * alpha[g] + ls;
    }
#pragma unroll
    for (int n = 0; n < 8; ++n)
#pragma unroll
      for (int g = 0; g < 8; ++g) o[n][g] *= alpha[g];

    // P (two f32 C tiles) -> bf16 A-operand fragment via LDS transpose
#pragma unroll
    for (int g = 0; g < 8; ++g) {
      plds[(rr + g) * 32 + coln]      = f2bf(p0[g]);
      plds[(rr + g) * 32 + 16 + coln] = f2bf(p1[g]);
    }
    asm volatile("s_wait_dscnt 0" ::: "memory");
    v16bf pf;
    {
      union { uint4 u[2]; v16bf v; } pu;
      const int r2 = lane & 15;
      pu.u[0] = *reinterpret_cast<const uint4*>(&plds[r2 * 32 + half * 16]);
      pu.u[1] = *reinterpret_cast<const uint4*>(&plds[r2 * 32 + half * 16 + 8]);
      pf = pu.v;
    }
    asm volatile("s_wait_dscnt 0" ::: "memory");

    const unsigned short* vb =
        Vp + ((((size_t)(b * KV_ + kvh) * (S_ / 32) + jj) * 8) * 512) + lane * 16;
#pragma unroll
    for (int n = 0; n < 8; ++n) o[n] = wmma_bf16(pf, ld_frag(vb + n * 512), o[n]);
  }

  float inv[8];
#pragma unroll
  for (int g = 0; g < 8; ++g) inv[g] = 1.f / lrow[g];
  const size_t rowbase = (size_t)(b * S_ + qi * 16 + rr) * (size_t)(H_ * HD_);
#pragma unroll
  for (int n = 0; n < 8; ++n) {
    const int col = h * HD_ + n * 16 + coln;
#pragma unroll
    for (int g = 0; g < 8; ++g)
      attnO[rowbase + (size_t)g * (H_ * HD_) + col] = o[n][g] * inv[g];
  }
}

// ---------------------------------------------------------------------------
extern "C" void kernel_launch(void* const* d_in, const int* in_sizes, int n_in,
                              void* d_out, int out_size, void* d_ws, size_t ws_size,
                              hipStream_t stream) {
  (void)in_sizes; (void)n_in; (void)out_size; (void)ws_size;
  const float* hidden = (const float*)d_in[0];
  const float* cosT   = (const float*)d_in[1];
  const float* sinT   = (const float*)d_in[2];
  // d_in[3] = attention_mask (pure causal; applied analytically in-kernel)
  const float* Wq = (const float*)d_in[4];
  const float* Wk = (const float*)d_in[5];
  const float* Wv = (const float*)d_in[6];
  const float* Wo = (const float*)d_in[7];
  const float* qw = (const float*)d_in[8];
  const float* kw = (const float*)d_in[9];
  float* out = (float*)d_out;

  char* ws = (char*)d_ws;
  size_t off = 0;
  auto take = [&](size_t bytes) {
    char* r = ws + off;
    off += (bytes + 255) & ~(size_t)255;
    return r;
  };
  unsigned short* hidA = (unsigned short*)take((size_t)M_ * D_ * 2);        // 16MB
  unsigned short* WqB  = (unsigned short*)take((size_t)D_ * H_ * HD_ * 2);  // 8MB
  unsigned short* WkB  = (unsigned short*)take((size_t)D_ * KV_ * HD_ * 2); // 4MB
  unsigned short* WvB  = (unsigned short*)take((size_t)D_ * KV_ * HD_ * 2); // 4MB
  unsigned short* WoB  = (unsigned short*)take((size_t)H_ * HD_ * D_ * 2);  // 8MB
  float* qf32 = (float*)take((size_t)M_ * H_ * HD_ * 4);                    // 32MB
  float* kf32 = (float*)take((size_t)M_ * KV_ * HD_ * 4);                   // 16MB
  float* vf32 = (float*)take((size_t)M_ * KV_ * HD_ * 4);                   // 16MB
  unsigned short* Qp = (unsigned short*)take((size_t)B_ * H_ * S_ * HD_ * 2);   // 16MB
  unsigned short* Kp = (unsigned short*)take((size_t)B_ * KV_ * S_ * HD_ * 2);  // 8MB
  unsigned short* Vp = (unsigned short*)take((size_t)B_ * KV_ * S_ * HD_ * 2);  // 8MB
  float* attnF = qf32;           // alias: q_f32 is dead after qnorm pass
  unsigned short* attnA = hidA;  // alias: hidden pack is dead after QKV GEMMs

  // 1) pack activations + weights into WMMA fragment layouts (bf16)
  pack_a_kernel<<<dim3((M_ / 16) * (D_ / 32) / 8), 256, 0, stream>>>(hidden, hidA, M_, D_);
  pack_b_kernel<<<dim3((D_ / 32) * (H_ * HD_ / 16) / 8), 256, 0, stream>>>(Wq, WqB, D_, H_ * HD_);
  pack_b_kernel<<<dim3((D_ / 32) * (KV_ * HD_ / 16) / 8), 256, 0, stream>>>(Wk, WkB, D_, KV_ * HD_);
  pack_b_kernel<<<dim3((D_ / 32) * (KV_ * HD_ / 16) / 8), 256, 0, stream>>>(Wv, WvB, D_, KV_ * HD_);
  pack_b_kernel<<<dim3((D_ / 32) * (H_ * HD_ / 16) / 8), 256, 0, stream>>>(Wo, WoB, H_ * HD_, D_);

  // 2) QKV projections (WMMA, double-buffered async-LDS staged B tiles)
  gemm_kernel<<<dim3(M_ / 128, (H_ * HD_) / 128), 256, 0, stream>>>(hidA, WqB, qf32, M_, H_ * HD_, D_);
  gemm_kernel<<<dim3(M_ / 128, (KV_ * HD_) / 128), 256, 0, stream>>>(hidA, WkB, kf32, M_, KV_ * HD_, D_);
  gemm_kernel<<<dim3(M_ / 128, (KV_ * HD_) / 128), 256, 0, stream>>>(hidA, WvB, vf32, M_, KV_ * HD_, D_);

  // 3) RMSNorm + RoPE + fragment-layout repack
  qnorm_rope_pack<<<dim3(B_ * S_ * H_ / 8), 256, 0, stream>>>(qf32, qw, cosT, sinT, Qp);
  knorm_rope_pack<<<dim3(B_ * S_ * KV_ / 8), 256, 0, stream>>>(kf32, kw, cosT, sinT, Kp);
  vpack_kernel<<<dim3(B_ * S_ * KV_ / 8), 256, 0, stream>>>(vf32, Vp);

  // 4) flash attention (causal, online softmax)
  flash_kernel<<<dim3(B_ * H_, S_ / 16), 32, 0, stream>>>(Qp, Kp, Vp, attnF);

  // 5) output projection
  pack_a_kernel<<<dim3((M_ / 16) * (D_ / 32) / 8), 256, 0, stream>>>(attnF, attnA, M_, H_ * HD_);
  gemm_kernel<<<dim3(M_ / 128, D_ / 128), 256, 0, stream>>>(attnA, WoB, out, M_, D_, H_ * HD_);
}
